// CTRNN_MD_32555852103837
// MI455X (gfx1250) — matrix operand order repaired and checked by reading, not compiled, not run on hardware
//
#include <hip/hip_runtime.h>
#include <hip/hip_bf16.h>

// ============================================================================
// CTRNN_MD on MI455X (gfx1250, wave32, WMMA, clusters, TDM)
//
//  - 8 clusters x 10 WGPs, 16 batch rows per cluster.
//  - WGP 0..7 : gate GEMMs for 64 h-columns each, v_wmma_f32_16x16x32_bf16
//  - WGP 8    : MD path (md_new GEMM + EMA + gumbel softmax + md@mul_gates)
//  - WGP 9    : out_{t-1} = relu(h_{t-1} @ Wh2r.T + b), overlapped with step t
//  - Per-step x_t / h_{t-1} tiles staged to LDS via TENSOR_LOAD_TO_LDS (TDM,
//    TENSORcnt) instead of cooperative copies.
//  - 2x s_cluster_barrier per step; h double-buffered (f32 state + bf16 GEMM
//    operand) in workspace; all weights pre-converted to bf16 (L2-resident).
// ============================================================================

#define T_STEPS 512
#define BATCH   128
#define IN_DIM  256
#define H_DIM   512
#define H2_DIM  1024
#define MD_DIM  16
#define OUT_DIM 64

#define NCLUST        8
#define WG_PER_CLUST  10   // 8 gate + 1 md + 1 out
#define ROWS_PER_CL   16

typedef __attribute__((ext_vector_type(16))) __bf16 v16bf;
typedef __attribute__((ext_vector_type(8)))  __bf16 v8bf;
typedef __attribute__((ext_vector_type(8)))  float  v8f;
typedef __attribute__((ext_vector_type(4)))  unsigned int v4u;
typedef __attribute__((ext_vector_type(8)))  int    v8i;
typedef __attribute__((ext_vector_type(4)))  int    v4i;

// ---------------------------------------------------------------------------
// A-fragment (16x32 bf16, ISA 7.12.2): lane L holds row (L&15);
//   elems 0..7  <-> K = kbase + 8*(L>>4) + 0..7
//   elems 8..15 <-> K = kbase + 16 + 8*(L>>4) + 0..7
__device__ __forceinline__ v16bf load_a_frag(const __bf16* rowp, int kbase, int laneHi) {
    v8bf lo = *(const v8bf*)(rowp + kbase + 8 * laneHi);
    v8bf hi = *(const v8bf*)(rowp + kbase + 16 + 8 * laneHi);
    return __builtin_shufflevector(lo, hi, 0,1,2,3,4,5,6,7,8,9,10,11,12,13,14,15);
}

// B-fragment (32x16 bf16): lane L holds column (L&15);
//   elems 0..7  <-> K = kbase + 8*(L>>4) + 0..7
//   elems 8..15 <-> K = kbase + 16 + 8*(L>>4) + 0..7
__device__ __forceinline__ v16bf load_b_frag(const __bf16* wrow, int kbase, int laneHi) {
    v8bf lo = *(const v8bf*)(wrow + kbase + 8 * laneHi);
    v8bf hi = *(const v8bf*)(wrow + kbase + 16 + 8 * laneHi);
    return __builtin_shufflevector(lo, hi, 0,1,2,3,4,5,6,7,8,9,10,11,12,13,14,15);
}

__device__ __forceinline__ v8f wmma_bf16(v16bf a, v16bf b, v8f c) {
    return __builtin_amdgcn_wmma_f32_16x16x32_bf16(false, a, false, b, (short)0, c,
                                                   false, false);
}

__device__ __forceinline__ float sigmoid_f(float x) { return 1.0f / (1.0f + __expf(-x)); }

__device__ __forceinline__ void cluster_sync() {
    __threadfence();                       // make L2-visible before cross-WGP sync
    __builtin_amdgcn_s_cluster_barrier();  // s_barrier_signal -3 ; s_barrier_wait -3
}

// Generic-pointer low 32 bits == LDS byte offset (ISA 10.2: LDS aperture).
__device__ __forceinline__ unsigned lds_off(const void* p) {
    return (unsigned)(unsigned long long)p;
}

// ---------------------------------------------------------------------------
// TDM: async DMA of a row-major 2D bf16 tile [tile_rows x row_elems] from
// global to LDS.  D# built per ISA 8.3-8.6; 2D => remaining groups zero.
// Tracked on TENSORcnt; pair with __builtin_amdgcn_s_wait_tensorcnt(0).
// (6-arg builtin form on this toolchain: g0, g1, g2, g3, g4, cpol.)
__device__ __forceinline__ void tdm_load_tile_bf16(unsigned lds_addr,
                                                   const void* gptr,
                                                   unsigned row_elems,   // tile&tensor dim0
                                                   unsigned tile_rows,   // tile dim1
                                                   unsigned row_stride)  // elems between rows
{
    unsigned long long ga = (unsigned long long)gptr;
    v4u g0;
    g0[0] = 1u;                                          // count=1, user descriptor
    g0[1] = lds_addr;                                    // lds_addr[31:0]
    g0[2] = (unsigned)(ga & 0xffffffffu);                // global_addr[31:0]
    g0[3] = (unsigned)((ga >> 32) & 0x01ffffffu)         // global_addr[56:32]
          | (2u << 30);                                  // type = 2 ("image")
    v8i g1;
    g1[0] = (int)(1u << 16);                             // data_size=1 (2B); no multicast
    g1[1] = (int)((row_elems & 0xffffu) << 16);          // tensor_dim0[15:0]
    g1[2] = (int)(((row_elems >> 16) & 0xffffu)          // tensor_dim0[31:16]
          |       ((tile_rows & 0xffffu) << 16));        // tensor_dim1[15:0]
    g1[3] = (int)(((tile_rows >> 16) & 0xffffu)          // tensor_dim1[31:16]
          |       ((row_elems & 0xffffu) << 16));        // tile_dim0
    g1[4] = (int)(tile_rows & 0xffffu);                  // tile_dim1 (tile_dim2=0)
    g1[5] = (int)row_stride;                             // tensor_dim0_stride[31:0]
    g1[6] = 0;                                           // stride[47:32] | dim1_stride lo
    g1[7] = 0;
    v4i gz4 = {0, 0, 0, 0};
    v8i gz8 = {0, 0, 0, 0, 0, 0, 0, 0};
    __builtin_amdgcn_tensor_load_to_lds(g0, g1, gz4, gz4, gz8, 0);
}

// ============================================================================
// Setup kernels (parallel, run once before the recurrent kernel)
// ============================================================================
__global__ void k_cvt_f32_to_bf16(const float* __restrict__ src,
                                  __bf16* __restrict__ dst, int n) {
    int i = blockIdx.x * blockDim.x + threadIdx.x;
    if (i < n) dst[i] = (__bf16)src[i];
}

// mulT[j][m] = mul_gates[m][j], padded along m to 32 with zeros (K=16 -> K=32 pad)
__global__ void k_build_mulT(const float* __restrict__ mul_gates,
                             __bf16* __restrict__ mulT) {
    int i = blockIdx.x * blockDim.x + threadIdx.x;  // over H_DIM*32
    if (i < H_DIM * 32) {
        int j = i >> 5, m = i & 31;
        mulT[i] = (m < MD_DIM) ? (__bf16)mul_gates[m * H_DIM + j] : (__bf16)0.0f;
    }
}

__global__ void k_init_state(const float* __restrict__ h0, const float* __restrict__ md0,
                             float* __restrict__ hF32, __bf16* __restrict__ hBF,
                             float* __restrict__ mdstate) {
    int i = blockIdx.x * blockDim.x + threadIdx.x;
    if (i < BATCH * H_DIM) { hF32[i] = h0[i]; hBF[i] = (__bf16)h0[i]; }
    if (i < BATCH * MD_DIM) { mdstate[i] = md0[i]; }
}

// ============================================================================
// Persistent recurrent kernel: one cluster of 10 WGPs per 16 batch rows.
// ============================================================================
__global__ void __launch_bounds__(128)
k_ctrnn_recurrent(const __bf16* __restrict__ xbf,          // [T][B][IN]
                  float*  __restrict__ hF32,               // [2][B][H]
                  __bf16* __restrict__ hBF,                // [2][B][H]
                  float*  __restrict__ mdstate,            // [B][MD]
                  float*  __restrict__ mg,                 // [B][H]
                  const __bf16* __restrict__ Wx2h_bf,      // [2H][IN]
                  const __bf16* __restrict__ Wh2h_bf,      // [2H][H]
                  const __bf16* __restrict__ Wx2md_bf,     // [MD][IN]
                  const __bf16* __restrict__ Wh2md_bf,     // [MD][H]
                  const __bf16* __restrict__ Wh2r_bf,      // [OUT][H]
                  const __bf16* __restrict__ mulT,         // [H][32] (m-padded)
                  const float* __restrict__ gumbel,        // [T][B][MD]
                  const float* __restrict__ bx2h,          // [2H]
                  const float* __restrict__ bh2h,          // [2H]
                  const float* __restrict__ bx2md,         // [MD]
                  const float* __restrict__ bh2md,         // [MD]
                  const float* __restrict__ bh2r,          // [OUT]
                  float* __restrict__ out)                 // outs | h | md
{
    const int cid    = blockIdx.x / WG_PER_CLUST;   // cluster id (0..7)
    const int wg     = blockIdx.x % WG_PER_CLUST;   // workgroup-in-cluster (0..9)
    const int b0     = cid * ROWS_PER_CL;           // batch-row base of cluster
    const int lane   = threadIdx.x & 31;
    const int laneLo = lane & 15;
    const int laneHi = lane >> 4;
    const int wave   = threadIdx.x >> 5;            // 4 waves per WGP

    // LDS staging of this step's A operands (shared by all 4 waves)
    __shared__ __align__(16) __bf16 sx[ROWS_PER_CL * IN_DIM];  //  8 KB
    __shared__ __align__(16) __bf16 sh[ROWS_PER_CL * H_DIM];   // 16 KB
    __shared__ __align__(16) __bf16 smd[ROWS_PER_CL * MD_DIM]; // 0.5 KB

    // Gate WGP: this wave owns 16 columns
    const int jt = wg * 4 + wave;        // tile id 0..31 (gate WGPs only)
    const int j  = jt * 16 + laneLo;     // h-column of this lane (0..511)

    for (int t = 0; t <= T_STEPS; ++t) {
        const int cur = t & 1;           // buffer holding h_{t-1}
        const int nxt = cur ^ 1;

        v8f c1 = {};   // keep logit  (x@Wx2h.T[:, j]   + h@Wh2h.T[:, j])
        v8f c2a = {};  // rec_in      (x@Wx2h.T[:, H+j])
        v8f c2b = {};  // ext_in      (h@Wh2h.T[:, H+j])

        // -------- stage x_t / h_{t-1} rows into LDS via TDM (TENSORcnt) ----
        const bool gateOrMd = (wg <= 8) && (t < T_STEPS);
        const bool outWork  = (wg == 9) && (t >= 1);
        if (wave == 0) {
            if (gateOrMd)
                tdm_load_tile_bf16(lds_off(sx),
                                   xbf + ((size_t)t * BATCH + b0) * IN_DIM,
                                   IN_DIM, ROWS_PER_CL, IN_DIM);
            if (gateOrMd || outWork)
                tdm_load_tile_bf16(lds_off(sh),
                                   hBF + (size_t)cur * BATCH * H_DIM
                                       + (size_t)b0 * H_DIM,
                                   H_DIM, ROWS_PER_CL, H_DIM);
            __builtin_amdgcn_s_wait_tensorcnt(0);
        }
        if (gateOrMd && t + 1 < T_STEPS)  // warm L2 for next step's x tile
            __builtin_prefetch(xbf + ((size_t)(t + 1) * BATCH + b0) * IN_DIM
                                   + (size_t)threadIdx.x * 32, 0, 1);
        __syncthreads();

        // ---------------- phase A: WMMA accumulation -----------------------
        if (gateOrMd && wg < 8) {
            const __bf16* xrow = sx + laneLo * IN_DIM;
            const __bf16* hrow = sh + laneLo * H_DIM;
            const __bf16* wxj  = Wx2h_bf + (size_t)j * IN_DIM;
            const __bf16* wxHj = Wx2h_bf + (size_t)(H_DIM + j) * IN_DIM;
            const __bf16* whj  = Wh2h_bf + (size_t)j * H_DIM;
            const __bf16* whHj = Wh2h_bf + (size_t)(H_DIM + j) * H_DIM;
            #pragma unroll 4
            for (int k = 0; k < IN_DIM; k += 32) {         // x-part: c1 + c2a
                v16bf a = load_a_frag(xrow, k, laneHi);
                c1  = wmma_bf16(a, load_b_frag(wxj,  k, laneHi), c1);
                c2a = wmma_bf16(a, load_b_frag(wxHj, k, laneHi), c2a);
            }
            #pragma unroll 4
            for (int k = 0; k < H_DIM; k += 32) {          // h-part: c1 + c2b
                v16bf a = load_a_frag(hrow, k, laneHi);
                c1  = wmma_bf16(a, load_b_frag(whj,  k, laneHi), c1);
                c2b = wmma_bf16(a, load_b_frag(whHj, k, laneHi), c2b);
            }
        } else if (gateOrMd && wg == 8) {
            // -------- MD path: md_new = relu([x|h] @ [Wx2md|Wh2md].T + b) --
            if (wave == 0) {
                v8f cm = {};
                const __bf16* xrow = sx + laneLo * IN_DIM;
                const __bf16* hrow = sh + laneLo * H_DIM;
                const int m = laneLo;  // md column of this lane
                for (int k = 0; k < IN_DIM; k += 32)
                    cm = wmma_bf16(load_a_frag(xrow, k, laneHi),
                                   load_b_frag(Wx2md_bf + (size_t)m * IN_DIM, k, laneHi), cm);
                for (int k = 0; k < H_DIM; k += 32)
                    cm = wmma_bf16(load_a_frag(hrow, k, laneHi),
                                   load_b_frag(Wh2md_bf + (size_t)m * H_DIM, k, laneHi), cm);
                // EMA + gumbel softmax over the 16 md columns (lanes of a half)
                #pragma unroll
                for (int v = 0; v < 8; ++v) {
                    int br = v + 8 * laneHi;              // local batch row
                    int b  = b0 + br;
                    float mdn = fmaxf(cm[v] + bx2md[m] + bh2md[m], 0.0f);
                    float mdv = 0.9f * mdstate[b * MD_DIM + m] + 0.1f * mdn;
                    float z   = mdv + gumbel[((size_t)t * BATCH + b) * MD_DIM + m];
                    float mx  = z;
                    mx = fmaxf(mx, __shfl_xor(mx, 1, 32));
                    mx = fmaxf(mx, __shfl_xor(mx, 2, 32));
                    mx = fmaxf(mx, __shfl_xor(mx, 4, 32));
                    mx = fmaxf(mx, __shfl_xor(mx, 8, 32));
                    float e = __expf(z - mx);
                    float s = e;
                    s += __shfl_xor(s, 1, 32);
                    s += __shfl_xor(s, 2, 32);
                    s += __shfl_xor(s, 4, 32);
                    s += __shfl_xor(s, 8, 32);
                    float p = e / s;
                    mdstate[b * MD_DIM + m] = p;          // carried state
                    smd[br * MD_DIM + m]    = (__bf16)p;  // A operand for mg
                }
            }
            __syncthreads();
            // -------- mg = md @ mul_gates  (K padded 16->32, zeros) --------
            #pragma unroll
            for (int i = 0; i < 8; ++i) {
                int jj = (wave * 8 + i) * 16 + laneLo;    // h-column
                v8bf lo = *(const v8bf*)(smd + laneLo * MD_DIM + 8 * laneHi);
                v8bf hz; for (int e = 0; e < 8; ++e) hz[e] = (__bf16)0.0f;
                v16bf a = __builtin_shufflevector(lo, hz,
                            0,1,2,3,4,5,6,7,8,9,10,11,12,13,14,15);
                v16bf bm = load_b_frag(mulT + (size_t)jj * 32, 0, laneHi);
                v8f acc = {};
                acc = wmma_bf16(a, bm, acc);
                #pragma unroll
                for (int v = 0; v < 8; ++v)
                    mg[(b0 + v + 8 * laneHi) * H_DIM + jj] = acc[v];
            }
        } else if (outWork) {
            // -------- out_{t-1} = relu(h_{t-1} @ Wh2r.T + b), overlapped ---
            const int n = wave * 16 + laneLo;             // out column
            const __bf16* hrow = sh + laneLo * H_DIM;
            v8f acc = {};
            #pragma unroll 4
            for (int k = 0; k < H_DIM; k += 32)
                acc = wmma_bf16(load_a_frag(hrow, k, laneHi),
                                load_b_frag(Wh2r_bf + (size_t)n * H_DIM, k, laneHi), acc);
            #pragma unroll
            for (int v = 0; v < 8; ++v) {
                int b = b0 + v + 8 * laneHi;
                out[((size_t)(t - 1) * BATCH + b) * OUT_DIM + n] =
                    fmaxf(acc[v] + bh2r[n], 0.0f);
            }
        }

        __syncthreads();
        cluster_sync();   // barrier 1: mg visible to gate WGPs

        // ---------------- phase B: gate epilogue, write h_t ----------------
        if (gateOrMd && wg < 8) {
            const float bkx = bx2h[j], bkh = bh2h[j];
            const float brx = bx2h[H_DIM + j], beh = bh2h[H_DIM + j];
            const float* hPrev = hF32 + (size_t)cur * BATCH * H_DIM;
            float*   hNf = hF32 + (size_t)nxt * BATCH * H_DIM;
            __bf16*  hNb = hBF  + (size_t)nxt * BATCH * H_DIM;
            #pragma unroll
            for (int v = 0; v < 8; ++v) {
                int b = b0 + v + 8 * laneHi;
                float keep = sigmoid_f(c1[v] + bkx + bkh);
                float rec  = (c2a[v] + brx) * mg[b * H_DIM + j];
                float ext  = c2b[v] + beh;
                float nh   = sigmoid_f(ext + rec);
                float hn   = keep * hPrev[b * H_DIM + j] + (1.0f - keep) * nh;
                hNf[b * H_DIM + j] = hn;
                hNb[b * H_DIM + j] = (__bf16)hn;
            }
        }

        __syncthreads();
        cluster_sync();   // barrier 2: h_t visible for step t+1
    }

    // -------- final outputs: h_{T-1} and md (each WGP writes what it owns) --
    const size_t OH = (size_t)T_STEPS * BATCH * OUT_DIM;
    if (wg < 8) {
        const float* hf = hF32 + (size_t)(T_STEPS & 1) * BATCH * H_DIM;
        for (int i = threadIdx.x; i < ROWS_PER_CL * 64; i += blockDim.x) {
            int r = i / 64, c = i % 64;
            int b = b0 + r, jj = wg * 64 + c;
            out[OH + (size_t)b * H_DIM + jj] = hf[b * H_DIM + jj];
        }
    } else if (wg == 8) {
        for (int i = threadIdx.x; i < ROWS_PER_CL * MD_DIM; i += blockDim.x) {
            int r = i / MD_DIM, m = i % MD_DIM;
            int b = b0 + r;
            out[OH + (size_t)BATCH * H_DIM + b * MD_DIM + m] = mdstate[b * MD_DIM + m];
        }
    }
}

// ============================================================================
// Host launcher
// ============================================================================
extern "C" void kernel_launch(void* const* d_in, const int* in_sizes, int n_in,
                              void* d_out, int out_size, void* d_ws, size_t ws_size,
                              hipStream_t stream) {
    (void)in_sizes; (void)n_in; (void)out_size; (void)ws_size;
    const float* x        = (const float*)d_in[0];
    const float* gumbel   = (const float*)d_in[1];
    const float* hidden0  = (const float*)d_in[2];
    const float* md0      = (const float*)d_in[3];
    /* d_in[4] task_id unused */
    const float* Wx2h     = (const float*)d_in[5];
    const float* bx2h     = (const float*)d_in[6];
    const float* Wh2h     = (const float*)d_in[7];
    const float* bh2h     = (const float*)d_in[8];
    const float* Wh2md    = (const float*)d_in[9];
    const float* bh2md    = (const float*)d_in[10];
    const float* Wx2md    = (const float*)d_in[11];
    const float* bx2md    = (const float*)d_in[12];
    const float* Wh2r     = (const float*)d_in[13];
    const float* bh2r     = (const float*)d_in[14];
    const float* mulg     = (const float*)d_in[15];
    float* out = (float*)d_out;

    // -------- workspace carve-out (256B aligned), ~36.5 MB total ----------
    char* p = (char*)d_ws;
    auto alloc = [&](size_t bytes) -> char* {
        char* r = p; p += (bytes + 255) & ~(size_t)255; return r;
    };
    __bf16* xbf      = (__bf16*)alloc((size_t)T_STEPS * BATCH * IN_DIM * 2);
    __bf16* Wx2h_bf  = (__bf16*)alloc((size_t)H2_DIM * IN_DIM * 2);
    __bf16* Wh2h_bf  = (__bf16*)alloc((size_t)H2_DIM * H_DIM * 2);
    __bf16* Wx2md_bf = (__bf16*)alloc((size_t)MD_DIM * IN_DIM * 2);
    __bf16* Wh2md_bf = (__bf16*)alloc((size_t)MD_DIM * H_DIM * 2);
    __bf16* Wh2r_bf  = (__bf16*)alloc((size_t)OUT_DIM * H_DIM * 2);
    __bf16* mulT     = (__bf16*)alloc((size_t)H_DIM * 32 * 2);
    float*  hF32     = (float*)alloc((size_t)2 * BATCH * H_DIM * 4);
    __bf16* hBF      = (__bf16*)alloc((size_t)2 * BATCH * H_DIM * 2);
    float*  mdstate  = (float*)alloc((size_t)BATCH * MD_DIM * 4);
    float*  mg       = (float*)alloc((size_t)BATCH * H_DIM * 4);

    const int TB = 256;
    auto blocks = [&](int n) { return (n + TB - 1) / TB; };

    // setup: bf16 conversions (parallel, once)
    k_cvt_f32_to_bf16<<<blocks(T_STEPS * BATCH * IN_DIM), TB, 0, stream>>>(x, xbf, T_STEPS * BATCH * IN_DIM);
    k_cvt_f32_to_bf16<<<blocks(H2_DIM * IN_DIM), TB, 0, stream>>>(Wx2h, Wx2h_bf, H2_DIM * IN_DIM);
    k_cvt_f32_to_bf16<<<blocks(H2_DIM * H_DIM), TB, 0, stream>>>(Wh2h, Wh2h_bf, H2_DIM * H_DIM);
    k_cvt_f32_to_bf16<<<blocks(MD_DIM * IN_DIM), TB, 0, stream>>>(Wx2md, Wx2md_bf, MD_DIM * IN_DIM);
    k_cvt_f32_to_bf16<<<blocks(MD_DIM * H_DIM), TB, 0, stream>>>(Wh2md, Wh2md_bf, MD_DIM * H_DIM);
    k_cvt_f32_to_bf16<<<blocks(OUT_DIM * H_DIM), TB, 0, stream>>>(Wh2r, Wh2r_bf, OUT_DIM * H_DIM);
    k_build_mulT<<<blocks(H_DIM * 32), TB, 0, stream>>>(mulg, mulT);
    k_init_state<<<blocks(BATCH * H_DIM), TB, 0, stream>>>(hidden0, md0, hF32, hBF, mdstate);

    // recurrent kernel: 8 clusters x 10 WGPs, 128 threads (4 waves) each.
    // (Cluster dims {10,1,1} would be set via extended launch attributes on a
    //  cluster-capable runtime; compile-only here.)
    dim3 grid(NCLUST * WG_PER_CLUST), block(128);
    k_ctrnn_recurrent<<<grid, block, 0, stream>>>(
        xbf, hF32, hBF, mdstate, mg,
        Wx2h_bf, Wh2h_bf, Wx2md_bf, Wh2md_bf, Wh2r_bf, mulT,
        gumbel, bx2h, bh2h, bx2md, bh2md, bh2r, out);
}
